// simpleLSTM_62096637166217
// MI455X (gfx1250) — matrix operand order repaired and checked by reading, not compiled, not run on hardware
//
#include <hip/hip_runtime.h>

typedef __attribute__((ext_vector_type(16))) _Float16 v16h_t;
typedef __attribute__((ext_vector_type(8)))  _Float16 h8_t;
typedef __attribute__((ext_vector_type(8)))  float    f32x8;
typedef __attribute__((ext_vector_type(4)))  float    f32x4;
typedef unsigned int u32x4 __attribute__((ext_vector_type(4)));
typedef int          i32x8 __attribute__((ext_vector_type(8)));
typedef int          i32x4 __attribute__((ext_vector_type(4)));

#define LSTM_B 4096
#define LSTM_T 512
#define LSTM_I 2
#define LSTM_H 64
#define GSTR 260    // gates LDS row stride in floats (banks + 16B align)
#define HSTR 72     // h LDS row stride in halves (64 h + pad, 16B align)
#define XROW 1028   // x-stage row stride in floats: 1024 data + 4 pad dwords

union AFrag { v16h_t v; f32x4 q[2]; };
union HPack { h8_t h; f32x4 q; };

#if __has_builtin(__builtin_amdgcn_tanhf)
__device__ __forceinline__ float fast_tanh(float x) {
  return __builtin_amdgcn_tanhf(x);              // v_tanh_f32: 1 trans op
}
__device__ __forceinline__ float fast_sig(float x) {
  return __builtin_fmaf(0.5f, __builtin_amdgcn_tanhf(0.5f * x), 0.5f);
}
#else
__device__ __forceinline__ float fast_sig(float x) {
  return __builtin_amdgcn_rcpf(1.0f + __builtin_amdgcn_exp2f(-1.442695041f * x));
}
__device__ __forceinline__ float fast_tanh(float x) {
  return 2.0f * __builtin_amdgcn_rcpf(1.0f + __builtin_amdgcn_exp2f(-2.885390082f * x)) - 1.0f;
}
#endif

__global__ __launch_bounds__(128, 1) void lstm_wmma_kernel(
    const float* __restrict__ x,
    const float* __restrict__ w_ih,
    const float* __restrict__ w_hh,
    const float* __restrict__ b_ih,
    const float* __restrict__ b_hh,
    const float* __restrict__ w_out,
    const float* __restrict__ b_out,
    float* __restrict__ out)
{
  __shared__ float    gates[16 * GSTR];        // 16.6 KB
  __shared__ _Float16 hbuf[16 * HSTR];         // 2.3 KB
  __shared__ float    xstage[16 * XROW];       // 64.3 KB: padded rows, 1 pad dword / 256

  const int tid  = threadIdx.x;
  const int lane = tid & 31;
  const int wave = tid >> 5;
  const int lhi  = lane >> 4;      // 0: lanes 0-15, 1: lanes 16-31
  const int lm   = lane & 15;
  const int b0   = blockIdx.x * 16;

  // ================= stage the whole x tile into LDS (once) =================
  // Tile: 16 rows x 1024 f32, global row stride 4KB, contiguous rows at
  // x + b0*1024. TDM pads LDS dest 1 dword per 256 dwords -> 4-bank row skew.
#if __has_builtin(__builtin_amdgcn_tensor_load_to_lds)
  if (tid < 32) {   // one wave issues the DMA; TENSORcnt is per-wave
    const unsigned long long gaddr =
        (unsigned long long)(uintptr_t)(x + (size_t)b0 * (LSTM_T * LSTM_I));
    const unsigned int ldsa = (unsigned int)(uintptr_t)(&xstage[0]);
    u32x4 g0;
    g0[0] = 1u;                                   // count=1, user desc
    g0[1] = ldsa;                                 // lds_addr (bytes)
    g0[2] = (unsigned int)gaddr;                  // global_addr[31:0]
    g0[3] = (unsigned int)((gaddr >> 32) & 0x01FFFFFFu) | (2u << 30); // [56:32], type=2
    i32x8 g1;
    g1[0] = (int)((2u << 16)        // data_size = 4B
                | (1u << 20)        // pad_enable
                | (7u << 22));      // pad_interval: 256 dwords (pad_amount=0 -> 1 dword)
    g1[1] = (int)((1024u & 0xFFFFu) << 16);       // tensor_dim0[15:0]
    g1[2] = (int)(16u << 16);                     // tensor_dim0[31:16]=0 | tensor_dim1=16
    g1[3] = (int)(1024u << 16);                   // tensor_dim1[31:16]=0 | tile_dim0=1024
    g1[4] = 16;                                   // tile_dim1=16, tile_dim2=0
    g1[5] = 1024;                                 // tensor_dim0_stride[31:0]
    g1[6] = 0;                                    // stride0[47:32] | dim1_stride lo
    g1[7] = 0;
    i32x4 gz4 = {0, 0, 0, 0};
    i32x8 gz8 = {0, 0, 0, 0, 0, 0, 0, 0};
    __builtin_amdgcn_tensor_load_to_lds(g0, g1, gz4, gz4, gz8, 0);
    __builtin_amdgcn_s_wait_tensorcnt(0);
  }
#else
  for (int idx = tid; idx < 16 * 1024; idx += 128) {
    const int xb = idx >> 10, e = idx & 1023;
    xstage[xb * XROW + e + (e >> 8)] = x[((size_t)(b0 + xb) << 10) + e];
  }
#endif

  // ---- zero the h LDS buffer (h(0) = 0) ----
  for (int i = tid; i < 16 * HSTR / 2; i += 128)
    ((unsigned int*)hbuf)[i] = 0u;

  // ---- persistent B fragments: columns n = wave*64 + nt*16 + lm ----
  // B layout (32x16 per frag): lanes 0-15 hold K rows kf*32+0..7,16..23,
  // lanes 16-31 hold K rows kf*32+8..15,24..31 (matches A table order).
  v16h_t bf[4][3];
  float  bsum[4];
#pragma unroll
  for (int nt = 0; nt < 4; ++nt) {
    const int n = wave * 64 + nt * 16 + lm;
    bsum[nt] = b_ih[n] + b_hh[n];
#pragma unroll
    for (int kf = 0; kf < 2; ++kf) {
      const int k0 = kf * 32 + lhi * 8;
      const float* src = w_hh + n * LSTM_H + k0;   // w_hh[n][k] contiguous in k
      v16h_t tv;
#pragma unroll
      for (int i = 0; i < 16; ++i) tv[i] = (_Float16)src[i + (i >> 3) * 8];
      bf[nt][kf] = tv;
    }
    // K-frag 2: rows 64..95 -> w_ih^T at K=64,65, zeros elsewhere
    v16h_t tv;
#pragma unroll
    for (int i = 0; i < 16; ++i) tv[i] = (_Float16)0.0f;
    if (lhi == 0) {
      tv[0] = (_Float16)w_ih[n * 2 + 0];
      tv[1] = (_Float16)w_ih[n * 2 + 1];
    }
    bf[nt][2] = tv;
  }

  float cst[8], hv[8];
#pragma unroll
  for (int j = 0; j < 8; ++j) { cst[j] = 0.0f; hv[j] = 0.0f; }

  const int eb = tid >> 3;          // batch row this thread updates
  const int ek = (tid & 7) * 8;     // 8 consecutive hidden indices

  __syncthreads();                  // x staged + h zeroed visible to all

  for (int t = 0; t < LSTM_T; ++t) {
    // ---- phase 1: gates = [h|x] @ [w_hh|w_ih]^T + bias via WMMA ----
    // A layout (16x32 f16): lane halves read K-chunks at +0/+16 (lanes<16)
    // or +8/+24 (lanes>=16); two ds_load_b128 per h K-frag.
    AFrag a[2];
    const _Float16* hrow = hbuf + lm * HSTR;
    const int c0 = lhi * 8;
#pragma unroll
    for (int kf = 0; kf < 2; ++kf) {
      a[kf].q[0] = *(const f32x4*)(const void*)(hrow + kf * 32 + c0);
      a[kf].q[1] = *(const f32x4*)(const void*)(hrow + kf * 32 + c0 + 16);
    }
    // x K-frag (K=64..95): lanes<16 hold K=64,65 in elements 0,1; rest zero.
    const float* xp = xstage + lm * XROW + t * 2 + (t >> 7);
    const float sx0 = lhi ? 0.0f : xp[0];
    const float sx1 = lhi ? 0.0f : xp[1];
    v16h_t a2;
#pragma unroll
    for (int i = 0; i < 16; ++i) a2[i] = (_Float16)0.0f;
    a2[0] = (_Float16)sx0;
    a2[1] = (_Float16)sx1;

#pragma unroll
    for (int nt = 0; nt < 4; ++nt) {
      f32x8 acc;
#pragma unroll
      for (int r = 0; r < 8; ++r) acc[r] = bsum[nt];   // bias folded into C
      acc = __builtin_amdgcn_wmma_f32_16x16x32_f16(
          false, a[0].v, false, bf[nt][0], (short)0, acc, false, false);
      acc = __builtin_amdgcn_wmma_f32_16x16x32_f16(
          false, a[1].v, false, bf[nt][1], (short)0, acc, false, false);
      acc = __builtin_amdgcn_wmma_f32_16x16x32_f16(
          false, a2,     false, bf[nt][2], (short)0, acc, false, false);
      // C layout: VGPR r -> row r (lanes 0-15) / row 8+r (lanes 16-31), col = lm
      const int ncol = wave * 64 + nt * 16 + lm;
#pragma unroll
      for (int r = 0; r < 8; ++r)
        gates[(r + lhi * 8) * GSTR + ncol] = acc[r];
    }
    __syncthreads();   // gates visible; A-frag reads of hbuf complete

    // ---- phase 2: activations + state update ----
    const float* grow = gates + eb * GSTR + ek;
#pragma unroll
    for (int j = 0; j < 8; ++j) {
      const float ii = fast_sig (grow[  0 + j]);
      const float ff = fast_sig (grow[ 64 + j]);
      const float gg = fast_tanh(grow[128 + j]);
      const float oo = fast_sig (grow[192 + j]);
      const float c  = ff * cst[j] + ii * gg;
      cst[j] = c;
      hv[j]  = oo * fast_tanh(c);
    }
    HPack hp;
#pragma unroll
    for (int j = 0; j < 8; ++j) hp.h[j] = (_Float16)hv[j];
    *(f32x4*)(void*)(hbuf + eb * HSTR + ek) = hp.q;   // one ds_store_b128
    __syncthreads();   // h(t) ready for next step
  }

  // ---- output projection: out = h_last @ w_out^T + b_out ----
#pragma unroll
  for (int j = 0; j < 8; ++j) gates[eb * GSTR + ek + j] = hv[j];  // f32 h_last
  __syncthreads();
  if (tid < 32) {
    const int ob = tid >> 1, oc = tid & 1;
    float s = b_out[oc];
    for (int k = 0; k < LSTM_H; ++k)
      s += gates[ob * GSTR + k] * w_out[oc * LSTM_H + k];
    out[(size_t)(b0 + ob) * 2 + oc] = s;
  }
}

extern "C" void kernel_launch(void* const* d_in, const int* in_sizes, int n_in,
                              void* d_out, int out_size, void* d_ws, size_t ws_size,
                              hipStream_t stream) {
  (void)in_sizes; (void)n_in; (void)out_size; (void)d_ws; (void)ws_size;
  const float* x     = (const float*)d_in[0];
  const float* w_ih  = (const float*)d_in[1];
  const float* w_hh  = (const float*)d_in[2];
  const float* b_ih  = (const float*)d_in[3];
  const float* b_hh  = (const float*)d_in[4];
  const float* w_out = (const float*)d_in[5];
  const float* b_out = (const float*)d_in[6];
  float* out = (float*)d_out;
  dim3 grid(LSTM_B / 16), block(128);
  hipLaunchKernelGGL(lstm_wmma_kernel, grid, block, 0, stream,
                     x, w_ih, w_hh, b_ih, b_hh, w_out, b_out, out);
}
